// TheirMonotonicLSTMCell_16810501997190
// MI455X (gfx1250) — matrix operand order repaired and checked by reading, not compiled, not run on hardware
//
#include <hip/hip_runtime.h>

typedef __attribute__((ext_vector_type(16))) _Float16 v16h;
typedef __attribute__((ext_vector_type(8)))  float    v8f;

#define NB   65536   // batch
#define ND   1024    // feature dim
#define GATE_STRIDE 34   // padded row stride (floats) in gate scratch (bank-conflict free)

__device__ __forceinline__ float hsig(float x) {
  return fminf(fmaxf(x * 0.16666667f + 0.5f, 0.0f), 1.0f);
}

// Load one lane's share of a 16x32 fp16 A-fragment from a row of x (fp32 in memory).
// p points at x_row + kchunk*32 + khi ; lane needs k = {khi..khi+7} U {16+khi..16+khi+7}.
__device__ __forceinline__ v16h load_a_frag(const float* p) {
  const float4* q = (const float4*)p;
  float4 r0 = q[0];   // k + 0..3
  float4 r1 = q[1];   // k + 4..7
  float4 r2 = q[4];   // k + 16..19
  float4 r3 = q[5];   // k + 20..23
  v16h a;
  a[0]=(_Float16)r0.x; a[1]=(_Float16)r0.y; a[2]=(_Float16)r0.z; a[3]=(_Float16)r0.w;
  a[4]=(_Float16)r1.x; a[5]=(_Float16)r1.y; a[6]=(_Float16)r1.z; a[7]=(_Float16)r1.w;
  a[8]=(_Float16)r2.x; a[9]=(_Float16)r2.y; a[10]=(_Float16)r2.z; a[11]=(_Float16)r2.w;
  a[12]=(_Float16)r3.x; a[13]=(_Float16)r3.y; a[14]=(_Float16)r3.z; a[15]=(_Float16)r3.w;
  return a;
}

__global__ __launch_bounds__(256)
void mono_lstm_fused(
    const float* __restrict__ x,   const float* __restrict__ hp,
    const float* __restrict__ cp,  const float* __restrict__ zp,
    const float* __restrict__ W_xi, const float* __restrict__ W_xf,
    const float* __restrict__ W_xc, const float* __restrict__ W_xo,
    const float* __restrict__ W_hi, const float* __restrict__ W_hf,
    const float* __restrict__ W_hc, const float* __restrict__ W_ho,
    const float* __restrict__ W_zi, const float* __restrict__ W_zf,
    const float* __restrict__ W_zc, const float* __restrict__ W_zo,
    const float* __restrict__ b_i,  const float* __restrict__ b_f,
    const float* __restrict__ b_c,  const float* __restrict__ b_o,
    const float* __restrict__ Wd1,  const float* __restrict__ bd1,
    const float* __restrict__ Wd2,  const float* __restrict__ bd2,
    const float* __restrict__ Wd3,  const float* __restrict__ bd3,
    const float* __restrict__ m_x,  const float* __restrict__ m_z,
    const float* __restrict__ m_1,  const float* __restrict__ m_2,
    float* __restrict__ out)
{
  extern __shared__ unsigned char smem[];
  // ---- LDS carve ----
  _Float16* sB   = (_Float16*)smem;                 // 32768 fp16 = 65536 B (WMMA-B fragments, all K)
  float* sGate   = (float*)(smem + 65536);          // 8 waves * 32 rows * 34 = 8704 floats
  float* sWh     = sGate + 8 * 32 * GATE_STRIDE;    // 256
  float* sZw     = sWh   + 256;                     // 32 : m_z[g]*W_zg[j]
  float* sBias   = sZw   + 32;                      // 32
  float* sWd1    = sBias + 32;                      // 40
  float* sBd1    = sWd1  + 40;                      // 5
  float* sWd2    = sBd1  + 5;                       // 25
  float* sBd2    = sWd2  + 25;                      // 5
  float* sWd3    = sBd2  + 5;                       // 5
  float* sBd3    = sWd3  + 5;                       // 1
  float* sM1     = sBd3  + 1;                       // 5
  float* sM2     = sM1   + 5;                       // 5

  const int tid = threadIdx.x;

  // ---- Prep: build mask-folded fp16 B fragments directly in WMMA-B lane order ----
  // fragment index i = kc*1024 + nt*512 + lane*16 + v
  //   col = nt*16 + (lane&15)  (gate g=col>>3, j=col&7)
  //   k   = kc*32 + (v>>3)*16 + (lane>>4)*8 + (v&7)
  for (int i = tid; i < 32768; i += 256) {
    int kc   = i >> 10;
    int nt   = (i >> 9) & 1;
    int lane = (i >> 4) & 31;
    int v    = i & 15;
    int col  = nt * 16 + (lane & 15);
    int g    = col >> 3;
    int j    = col & 7;
    int k    = kc * 32 + ((v >> 3) << 4) + ((lane >> 4) << 3) + (v & 7);
    const float* Wx = (g == 0) ? W_xi : (g == 1) ? W_xf : (g == 2) ? W_xc : W_xo;
    sB[i] = (_Float16)(m_x[g * ND + k] * Wx[k * 8 + j]);
  }
  {
    int g = tid >> 6, idx = tid & 63;
    const float* Wh = (g == 0) ? W_hi : (g == 1) ? W_hf : (g == 2) ? W_hc : W_ho;
    sWh[tid] = Wh[idx];
  }
  if (tid < 32) {
    int g = tid >> 3, j = tid & 7;
    const float* Wz = (g == 0) ? W_zi : (g == 1) ? W_zf : (g == 2) ? W_zc : W_zo;
    const float* bb = (g == 0) ? b_i  : (g == 1) ? b_f  : (g == 2) ? b_c  : b_o;
    sZw[tid]   = m_z[g] * Wz[j];
    sBias[tid] = bb[j];
  }
  if (tid < 40) sWd1[tid] = Wd1[tid];
  if (tid < 25) sWd2[tid] = Wd2[tid];
  if (tid < 5) {
    sBd1[tid] = bd1[tid]; sBd2[tid] = bd2[tid];
    sWd3[tid] = Wd3[tid]; sM1[tid]  = m_1[tid]; sM2[tid] = m_2[tid];
  }
  if (tid == 0) sBd3[0] = bd3[0];
  __syncthreads();

  // ---- GEMM: each wave owns a 32-row strip; x @ Wc[1024,32] via f16 WMMA, f32 accum ----
  const int wave    = tid >> 5;
  const int lane    = tid & 31;
  const int rowBase = blockIdx.x * 256 + wave * 32;
  const int half    = lane >> 4;     // 0/1
  const int m15     = lane & 15;
  const int khi     = half * 8;

  const float* xr0 = x + (size_t)(rowBase + m15) * ND;        // M-tile 0
  const float* xr1 = x + (size_t)(rowBase + 16 + m15) * ND;   // M-tile 1

  v8f acc00 = {}, acc01 = {}, acc10 = {}, acc11 = {};

  #pragma unroll 2
  for (int kc = 0; kc < 32; ++kc) {
    const int kb = kc * 32;
    v16h a0 = load_a_frag(xr0 + kb + khi);
    v16h a1 = load_a_frag(xr1 + kb + khi);
    v16h b0 = *(const v16h*)(sB + kc * 1024 + 0   + lane * 16);
    v16h b1 = *(const v16h*)(sB + kc * 1024 + 512 + lane * 16);
    acc00 = __builtin_amdgcn_wmma_f32_16x16x32_f16(false, a0, false, b0, (short)0, acc00, false, false);
    acc01 = __builtin_amdgcn_wmma_f32_16x16x32_f16(false, a0, false, b1, (short)0, acc01, false, false);
    acc10 = __builtin_amdgcn_wmma_f32_16x16x32_f16(false, a1, false, b0, (short)0, acc10, false, false);
    acc11 = __builtin_amdgcn_wmma_f32_16x16x32_f16(false, a1, false, b1, (short)0, acc11, false, false);
  }

  // ---- Spill C fragments to padded LDS so each lane owns one full row ----
  float* sg = sGate + wave * 32 * GATE_STRIDE;
  #pragma unroll
  for (int v = 0; v < 8; ++v) {
    int r0 = half * 8 + v;                               // C layout: M = v + (lane>>4)*8, N = lane&15
    sg[r0 * GATE_STRIDE + m15]             = acc00[v];
    sg[r0 * GATE_STRIDE + 16 + m15]        = acc01[v];
    sg[(16 + r0) * GATE_STRIDE + m15]      = acc10[v];
    sg[(16 + r0) * GATE_STRIDE + 16 + m15] = acc11[v];
  }
  __syncthreads();

  // ---- Per-row tail: recurrent H=8 part, activations, MLP head, stores ----
  const int r = rowBase + lane;
  const float* sgr = sg + lane * GATE_STRIDE;

  float4 h01 = *(const float4*)(hp + (size_t)r * 8);
  float4 h23 = *(const float4*)(hp + (size_t)r * 8 + 4);
  float4 c01 = *(const float4*)(cp + (size_t)r * 8);
  float4 c23 = *(const float4*)(cp + (size_t)r * 8 + 4);
  float h[8]     = {h01.x, h01.y, h01.z, h01.w, h23.x, h23.y, h23.z, h23.w};
  float cprev[8] = {c01.x, c01.y, c01.z, c01.w, c23.x, c23.y, c23.z, c23.w};
  float z = zp[r];

  float g4[4][8];
  #pragma unroll
  for (int g = 0; g < 4; ++g) {
    #pragma unroll
    for (int j = 0; j < 8; ++j) {
      float val = sgr[g * 8 + j] + sBias[g * 8 + j] + z * sZw[g * 8 + j];
      #pragma unroll
      for (int i = 0; i < 8; ++i) val += h[i] * sWh[g * 64 + i * 8 + j];
      g4[g][j] = val;
    }
  }

  float ht_[8], ct_[8];
  #pragma unroll
  for (int j = 0; j < 8; ++j) {
    float It = hsig(g4[0][j]);
    float Ft = hsig(g4[1][j]);
    float cc = Ft * cprev[j] + It * tanhf(g4[2][j]);
    float Ot = hsig(g4[3][j]);
    ct_[j] = cc;
    ht_[j] = Ot * tanhf(cc);
  }

  float d1[5];
  #pragma unroll
  for (int k = 0; k < 5; ++k) {
    float s = sBd1[k];
    #pragma unroll
    for (int j = 0; j < 8; ++j) s += ht_[j] * sWd1[j * 5 + k];
    d1[k] = fmaxf(s, 0.0f) * sM1[k];
  }
  float d2[5];
  #pragma unroll
  for (int k = 0; k < 5; ++k) {
    float s = sBd2[k];
    #pragma unroll
    for (int j = 0; j < 5; ++j) s += d1[j] * sWd2[j * 5 + k];
    d2[k] = fmaxf(s, 0.0f) * sM2[k];
  }
  float s3 = sBd3[0];
  #pragma unroll
  for (int j = 0; j < 5; ++j) s3 += d2[j] * sWd3[j];
  float dt = fmaxf(s3, 0.0f);

  // outputs: zt [B] | ht [B,8] | ct [B,8]
  out[r] = z + dt;
  float4* oh = (float4*)(out + NB + (size_t)r * 8);
  oh[0] = make_float4(ht_[0], ht_[1], ht_[2], ht_[3]);
  oh[1] = make_float4(ht_[4], ht_[5], ht_[6], ht_[7]);
  float4* oc = (float4*)(out + NB + (size_t)8 * NB + (size_t)r * 8);
  oc[0] = make_float4(ct_[0], ct_[1], ct_[2], ct_[3]);
  oc[1] = make_float4(ct_[4], ct_[5], ct_[6], ct_[7]);
}

extern "C" void kernel_launch(void* const* d_in, const int* in_sizes, int n_in,
                              void* d_out, int out_size, void* d_ws, size_t ws_size,
                              hipStream_t stream) {
  const float* x    = (const float*)d_in[0];
  const float* hp   = (const float*)d_in[1];
  const float* cp   = (const float*)d_in[2];
  const float* zp   = (const float*)d_in[3];
  const float* W_xi = (const float*)d_in[4];
  const float* W_xf = (const float*)d_in[5];
  const float* W_xc = (const float*)d_in[6];
  const float* W_xo = (const float*)d_in[7];
  const float* W_hi = (const float*)d_in[8];
  const float* W_hf = (const float*)d_in[9];
  const float* W_hc = (const float*)d_in[10];
  const float* W_ho = (const float*)d_in[11];
  const float* W_zi = (const float*)d_in[12];
  const float* W_zf = (const float*)d_in[13];
  const float* W_zc = (const float*)d_in[14];
  const float* W_zo = (const float*)d_in[15];
  const float* b_i  = (const float*)d_in[16];
  const float* b_f  = (const float*)d_in[17];
  const float* b_c  = (const float*)d_in[18];
  const float* b_o  = (const float*)d_in[19];
  const float* Wd1  = (const float*)d_in[20];
  const float* bd1  = (const float*)d_in[21];
  const float* Wd2  = (const float*)d_in[22];
  const float* bd2  = (const float*)d_in[23];
  const float* Wd3  = (const float*)d_in[24];
  const float* bd3  = (const float*)d_in[25];
  const float* m_x  = (const float*)d_in[26];
  const float* m_z  = (const float*)d_in[27];
  const float* m_1  = (const float*)d_in[28];
  const float* m_2  = (const float*)d_in[29];
  float* out = (float*)d_out;

  // LDS: 64KB B-fragments + 34KB gate scratch + ~1.7KB tail constants
  size_t shmem = 65536
               + (size_t)(8 * 32 * GATE_STRIDE) * 4
               + (size_t)(256 + 32 + 32 + 40 + 5 + 25 + 5 + 5 + 1 + 5 + 5) * 4;

  dim3 grid(NB / 256), block(256);
  mono_lstm_fused<<<grid, block, shmem, stream>>>(
      x, hp, cp, zp,
      W_xi, W_xf, W_xc, W_xo,
      W_hi, W_hf, W_hc, W_ho,
      W_zi, W_zf, W_zc, W_zo,
      b_i, b_f, b_c, b_o,
      Wd1, bd1, Wd2, bd2, Wd3, bd3,
      m_x, m_z, m_1, m_2, out);
}
